// ElmanLayer_2147483648048
// MI455X (gfx1250) — matrix operand order repaired and checked by reading, not compiled, not run on hardware
//
#include <hip/hip_runtime.h>
#include <hip/hip_bf16.h>

typedef __attribute__((ext_vector_type(16))) _Float16 v16h;
typedef __attribute__((ext_vector_type(8)))  _Float16 v8h;
typedef __attribute__((ext_vector_type(8)))  float    v8f;

#define DIMD  1280
#define BATCH 16
#define SEQ   2048
#define MTOT  (BATCH*SEQ)            // 32768 rows
#define NX    ((size_t)MTOT*DIMD)    // 41,943,040 elements
#define NWIN  ((size_t)2*DIMD*DIMD)  // 3,276,800
#define NW    ((size_t)DIMD*DIMD)    // 1,638,400
#define NH    ((size_t)BATCH*DIMD)   // 20,480

__device__ __forceinline__ float silu_f(float x) {
    return x / (1.0f + __expf(-x));
}

// ---- WMMA fragment loads (f16, 16x16x32) -------------------------------
// A fragment: row-major 16xK tile. Lane L<16: row L, halves0..15 = K+0..15.
// Lane L>=16: row L-16, K+16..31.
__device__ __forceinline__ v16h load_a_frag(const _Float16* __restrict__ arow,
                                            int kk, int hsel) {
    v16h a = *(const v16h*)(arow + kk + hsel * 16);
    return a;
}
// B fragment: B[k][n] = W[n][k], W row-major NxK. Lane L<16: col L, K+0..15
// contiguous from W row; lane L>=16: col L-16, K+16..31.
__device__ __forceinline__ v16h load_b_frag(const _Float16* __restrict__ brow,
                                            int kk, int hsel) {
    return *(const v16h*)(brow + kk + hsel * 16);
}

// ---- CDNA5 async global->LDS copy (ASYNCcnt-tracked) -------------------
// LDS dest = LDS base + per-lane VGPR byte offset + imm offset; per-lane addresses.
__device__ __forceinline__ void async_copy_b128(unsigned lds_off,
                                                const void* gptr) {
    asm volatile("global_load_async_to_lds_b128 %0, %1, off"
                 :: "v"(lds_off), "v"(gptr) : "memory");
}
__device__ __forceinline__ void wait_asynccnt0() {
    asm volatile("s_wait_asynccnt 0" ::: "memory");
}

// ---- fp32 -> fp16 streaming convert ------------------------------------
__global__ void k_f32_to_f16(const float* __restrict__ src,
                             _Float16* __restrict__ dst, size_t n) {
    size_t i = (size_t)blockIdx.x * blockDim.x + threadIdx.x;
    if (i < n) dst[i] = (_Float16)src[i];
}

// ---- init: zero h double-buffer + sync counters ------------------------
__global__ void k_init(_Float16* __restrict__ h16, unsigned* __restrict__ sync) {
    size_t i = (size_t)blockIdx.x * blockDim.x + threadIdx.x;
    if (i < 2 * NH) h16[i] = (_Float16)0.0f;
    if (i < 4) sync[i] = 0u;
}

// ---- GEMM 1: xz = x16 @ W_in16^T, split + SiLU epilogue ----------------
// Each wave: one 16-row tile x 64 output cols (4 WMMA tiles), K = 1280.
__global__ __launch_bounds__(256)
void k_gemm_in(const _Float16* __restrict__ x16,
               const _Float16* __restrict__ Win16,
               _Float16* __restrict__ sx16,
               _Float16* __restrict__ sz16) {
    const int NSET = (2 * DIMD) / 64; // 40
    int wave = (int)((blockIdx.x * blockDim.x + threadIdx.x) >> 5);
    int lane = threadIdx.x & 31;
    int hsel = lane >> 4;
    int r    = lane & 15;
    int mt = wave / NSET;
    int ns = wave % NSET;
    if (mt >= MTOT / 16) return;

    const _Float16* arow  = x16   + (size_t)(mt * 16 + r) * DIMD;
    const _Float16* brow0 = Win16 + (size_t)(ns * 64 + r) * DIMD;
    v8f acc[4] = {};
    for (int kk = 0; kk < DIMD; kk += 32) {
        __builtin_prefetch((const void*)(arow + kk + 256), 0, 1);
        v16h a = load_a_frag(arow, kk, hsel);
#pragma unroll
        for (int j = 0; j < 4; ++j) {
            v16h b = load_b_frag(brow0 + (size_t)j * 16 * DIMD, kk, hsel);
            acc[j] = __builtin_amdgcn_wmma_f32_16x16x32_f16(
                false, a, false, b, (short)0, acc[j], false, false);
        }
    }
#pragma unroll
    for (int j = 0; j < 4; ++j) {
        int n = ns * 64 + j * 16 + r;
#pragma unroll
        for (int rr = 0; rr < 8; ++rr) {
            int row = rr + hsel * 8;
            size_t m = (size_t)(mt * 16 + row);
            float v = silu_f(acc[j][rr]);
            if (n < DIMD) sx16[m * DIMD + n]          = (_Float16)v;
            else          sz16[m * DIMD + (n - DIMD)] = (_Float16)v;
        }
    }
}

// ---- GEMM 2: x_pre = sx16 @ W_x16^T + b, stored f32 in (T,B,D) ---------
__global__ __launch_bounds__(256)
void k_gemm_xpre(const _Float16* __restrict__ sx16,
                 const _Float16* __restrict__ Wx16,
                 const float* __restrict__ bias,
                 float* __restrict__ xpre) {
    const int NSET = DIMD / 64; // 20
    int wave = (int)((blockIdx.x * blockDim.x + threadIdx.x) >> 5);
    int lane = threadIdx.x & 31;
    int hsel = lane >> 4;
    int r    = lane & 15;
    int mt = wave / NSET;
    int ns = wave % NSET;
    if (mt >= MTOT / 16) return;

    const _Float16* arow  = sx16 + (size_t)(mt * 16 + r) * DIMD;
    const _Float16* brow0 = Wx16 + (size_t)(ns * 64 + r) * DIMD;
    v8f acc[4] = {};
    for (int kk = 0; kk < DIMD; kk += 32) {
        __builtin_prefetch((const void*)(arow + kk + 256), 0, 1);
        v16h a = load_a_frag(arow, kk, hsel);
#pragma unroll
        for (int j = 0; j < 4; ++j) {
            v16h b = load_b_frag(brow0 + (size_t)j * 16 * DIMD, kk, hsel);
            acc[j] = __builtin_amdgcn_wmma_f32_16x16x32_f16(
                false, a, false, b, (short)0, acc[j], false, false);
        }
    }
#pragma unroll
    for (int j = 0; j < 4; ++j) {
        int n = ns * 64 + j * 16 + r;
        float bb = bias[n];
#pragma unroll
        for (int rr = 0; rr < 8; ++rr) {
            int row = rr + hsel * 8;
            int m = mt * 16 + row;        // m = b*SEQ + t
            int bi = m >> 11;             // /2048
            int t  = m & (SEQ - 1);
            xpre[((size_t)t * BATCH + bi) * DIMD + n] = acc[j][rr] + bb;
        }
    }
}

// ---- grid-wide barrier for the persistent scan kernel ------------------
__device__ __forceinline__ void grid_sync(unsigned* cnt, unsigned* gen,
                                          unsigned nb, unsigned target) {
    __syncthreads();
    if (threadIdx.x == 0) {
        __threadfence();
        unsigned old = atomicAdd(cnt, 1u);
        if (old == nb - 1) {
            __hip_atomic_store(cnt, 0u, __ATOMIC_RELAXED, __HIP_MEMORY_SCOPE_AGENT);
            __hip_atomic_fetch_add(gen, 1u, __ATOMIC_RELEASE, __HIP_MEMORY_SCOPE_AGENT);
        } else {
            while (__hip_atomic_load(gen, __ATOMIC_ACQUIRE,
                                     __HIP_MEMORY_SCOPE_AGENT) < target) {}
        }
    }
    __syncthreads();
}

// ---- Sequential Elman scan: persistent, 20 blocks x 4 waves = 80 tiles --
// Block stages its 64 W_h rows (160 KB) into LDS once via CDNA5 async
// global->LDS copies; per step each wave runs 40 WMMAs with B fragments
// served from LDS (ds_load) and h fragments from L2. One grid barrier/step.
__global__ __launch_bounds__(128)
void k_scan(const _Float16* __restrict__ Wh16,
            const float* __restrict__ xpre,
            const _Float16* __restrict__ sz16,
            _Float16* __restrict__ h16,     // 2 buffers of 16*DIMD
            float* __restrict__ h32,
            _Float16* __restrict__ out16,
            unsigned* __restrict__ sync) {
    extern __shared__ _Float16 smem[];    // 64 rows x DIMD halves = 160 KB
    const unsigned NB = 20;
    int wave = threadIdx.x >> 5;          // 0..3
    int lane = threadIdx.x & 31;
    int hsel = lane >> 4;
    int r    = lane & 15;
    int tile = blockIdx.x * 4 + wave;     // 0..79
    int n0   = tile * 16;
    int n    = n0 + r;

    // --- stage W_h rows [blockIdx*64, +64) into LDS (async, once) -------
    {
        const char* gbase = (const char*)(Wh16 + (size_t)blockIdx.x * 64 * DIMD);
        // low 32 bits of a generic LDS pointer == LDS byte offset (AS3 cast)
        unsigned lbase = (unsigned)(uintptr_t)(void*)smem;
        const int total = 64 * DIMD * 2;             // 163840 bytes
        for (int i = threadIdx.x * 16; i < total; i += 128 * 16) {
            async_copy_b128(lbase + (unsigned)i, gbase + i);
        }
        wait_asynccnt0();
        __syncthreads();
    }
    const _Float16* brow = smem + (size_t)(wave * 16 + r) * DIMD;

    unsigned target = 0;
    for (int t = 0; t < SEQ; ++t) {
        const _Float16* hbuf  = h16 + (size_t)(t & 1) * (BATCH * DIMD);
        _Float16*       hnext = h16 + (size_t)((t + 1) & 1) * (BATCH * DIMD);
        const _Float16* arow  = hbuf + (size_t)r * DIMD;

        v8f acc = {};
#pragma unroll 4
        for (int kk = 0; kk < DIMD; kk += 32) {
            v16h a = load_a_frag(arow, kk, hsel);
            v16h b = *(const v16h*)(brow + kk + hsel * 16);  // ds_load from LDS
            acc = __builtin_amdgcn_wmma_f32_16x16x32_f16(
                false, a, false, b, (short)0, acc, false, false);
        }
#pragma unroll
        for (int rr = 0; rr < 8; ++rr) {
            int row = rr + hsel * 8;      // batch index 0..15
            float xp = xpre[((size_t)t * BATCH + row) * DIMD + n];
            float h  = tanhf(xp + acc[rr]);
            hnext[(size_t)row * DIMD + n] = (_Float16)h;
            h32[(size_t)row * DIMD + n]   = h;
            size_t mi = ((size_t)row * SEQ + t) * DIMD + n;
            out16[mi] = (_Float16)(h * (float)sz16[mi]);
        }
        ++target;
        grid_sync(sync, sync + 1, NB, target);
    }
}

// ---- GEMM 3: out = out16 @ W_out16^T -> f32 ----------------------------
__global__ __launch_bounds__(256)
void k_gemm_out(const _Float16* __restrict__ out16,
                const _Float16* __restrict__ Wout16,
                float* __restrict__ out) {
    const int NSET = DIMD / 64; // 20
    int wave = (int)((blockIdx.x * blockDim.x + threadIdx.x) >> 5);
    int lane = threadIdx.x & 31;
    int hsel = lane >> 4;
    int r    = lane & 15;
    int mt = wave / NSET;
    int ns = wave % NSET;
    if (mt >= MTOT / 16) return;

    const _Float16* arow  = out16  + (size_t)(mt * 16 + r) * DIMD;
    const _Float16* brow0 = Wout16 + (size_t)(ns * 64 + r) * DIMD;
    v8f acc[4] = {};
    for (int kk = 0; kk < DIMD; kk += 32) {
        __builtin_prefetch((const void*)(arow + kk + 256), 0, 1);
        v16h a = load_a_frag(arow, kk, hsel);
#pragma unroll
        for (int j = 0; j < 4; ++j) {
            v16h b = load_b_frag(brow0 + (size_t)j * 16 * DIMD, kk, hsel);
            acc[j] = __builtin_amdgcn_wmma_f32_16x16x32_f16(
                false, a, false, b, (short)0, acc[j], false, false);
        }
    }
#pragma unroll
    for (int j = 0; j < 4; ++j) {
        int n = ns * 64 + j * 16 + r;
#pragma unroll
        for (int rr = 0; rr < 8; ++rr) {
            int row = rr + hsel * 8;
            size_t m = (size_t)(mt * 16 + row);
            out[m * DIMD + n] = acc[j][rr];
        }
    }
}

// ---- copy final hidden state to output tail ----------------------------
__global__ void k_copy_h(const float* __restrict__ h32, float* __restrict__ dst) {
    int i = blockIdx.x * blockDim.x + threadIdx.x;
    if (i < (int)NH) dst[i] = h32[i];
}

extern "C" void kernel_launch(void* const* d_in, const int* in_sizes, int n_in,
                              void* d_out, int out_size, void* d_ws, size_t ws_size,
                              hipStream_t stream) {
    const float* x     = (const float*)d_in[0];
    const float* W_in  = (const float*)d_in[1];
    const float* W_out = (const float*)d_in[2];
    const float* W_x   = (const float*)d_in[3];
    const float* W_h   = (const float*)d_in[4];
    const float* bias  = (const float*)d_in[5];
    float* out = (float*)d_out;

    char* ws = (char*)d_ws;
    size_t off = 0;
    auto alloc = [&](size_t bytes) -> char* {
        char* p = ws + off;
        off = (off + bytes + 255) & ~(size_t)255;
        return p;
    };
    _Float16* x16    = (_Float16*)alloc(NX * 2);
    _Float16* Win16  = (_Float16*)alloc(NWIN * 2);
    _Float16* Wx16   = (_Float16*)alloc(NW * 2);
    _Float16* Wh16   = (_Float16*)alloc(NW * 2);
    _Float16* Wout16 = (_Float16*)alloc(NW * 2);
    _Float16* sx16   = (_Float16*)alloc(NX * 2);
    _Float16* sz16   = (_Float16*)alloc(NX * 2);
    float*    xpre   = (float*)   alloc(NX * 4);
    _Float16* out16  = (_Float16*)alloc(NX * 2);
    _Float16* h16    = (_Float16*)alloc(2 * NH * 2);
    float*    h32    = (float*)   alloc(NH * 4);
    unsigned* sync   = (unsigned*)alloc(256);

    // 1) precision conversion passes (streaming, ~250 MB total @ 23.3 TB/s)
    k_f32_to_f16<<<(unsigned)((NX   + 255) / 256), 256, 0, stream>>>(x,     x16,    NX);
    k_f32_to_f16<<<(unsigned)((NWIN + 255) / 256), 256, 0, stream>>>(W_in,  Win16,  NWIN);
    k_f32_to_f16<<<(unsigned)((NW   + 255) / 256), 256, 0, stream>>>(W_x,   Wx16,   NW);
    k_f32_to_f16<<<(unsigned)((NW   + 255) / 256), 256, 0, stream>>>(W_h,   Wh16,   NW);
    k_f32_to_f16<<<(unsigned)((NW   + 255) / 256), 256, 0, stream>>>(W_out, Wout16, NW);
    k_init<<<(unsigned)((2 * NH + 255) / 256), 256, 0, stream>>>(h16, sync);

    // 2) input projection + SiLU split: 215 GFLOP via f16 WMMA
    k_gemm_in<<<(MTOT / 16) * 40 / 8, 256, 0, stream>>>(x16, Win16, sx16, sz16);
    // 3) recurrent-input projection + bias, laid out (T,B,D)
    k_gemm_xpre<<<(MTOT / 16) * 20 / 8, 256, 0, stream>>>(sx16, Wx16, bias, xpre);
    // 4) sequential tanh scan: persistent grid, W_h LDS-resident (160 KB)
    k_scan<<<20, 128, 64 * DIMD * 2, stream>>>(Wh16, xpre, sz16, h16, h32,
                                               out16, sync);
    // 5) output projection
    k_gemm_out<<<(MTOT / 16) * 20 / 8, 256, 0, stream>>>(out16, Wout16, out);
    // 6) h_final tail
    k_copy_h<<<(unsigned)((NH + 255) / 256), 256, 0, stream>>>(h32, out + NX);
}
